// WLSubtreeLayer_17446157156477
// MI455X (gfx1250) — matrix operand order repaired and checked by reading, not compiled, not run on hardware
//
#include <hip/hip_runtime.h>
#include <hip/hip_bf16.h>
#include <math.h>

typedef __attribute__((ext_vector_type(16))) __bf16 v16bf;
typedef __attribute__((ext_vector_type(8)))  float  v8f;
typedef __attribute__((ext_vector_type(2)))  int    v2i;
typedef __attribute__((ext_vector_type(4)))  int    v4i;

#define NN      50000
#define EE      800000
#define HIDK    512
#define MM      256
#define NSTEPS  3

#define TILE_R  32
#define LDA     40      // bf16 elems, padded (80 B rows, 16B aligned)
#define LDB     40      // bf16 elems, padded
#define LDCE    260     // f32 elems, padded
#define GBLK    ((NN + TILE_R - 1) / TILE_R)   // 1563

#define STG_BYTES ((TILE_R * LDA + 256 * LDB) * 2)      // 23040 B per staging buffer
#define SMEM_BYTES (2 * STG_BYTES)                      // 46080 B (C tile 33280 B aliases)

// ---- async global->LDS path (CDNA5), guarded so fallback always compiles ----
#if defined(__has_builtin)
#if __has_builtin(__builtin_amdgcn_global_load_async_to_lds_b64) && \
    __has_builtin(__builtin_amdgcn_global_load_async_to_lds_b128)
#define HAVE_ASYNC_LDS 1
#endif
#endif
#ifndef HAVE_ASYNC_LDS
#define HAVE_ASYNC_LDS 0
#endif

#define AS1 __attribute__((address_space(1)))
#define AS3 __attribute__((address_space(3)))

#if HAVE_ASYNC_LDS
__device__ __forceinline__ AS1 v2i* gp64(const void* p) {
    return (AS1 v2i*)(unsigned long long)p;
}
__device__ __forceinline__ AS3 v2i* sp64(void* p) {
    return (AS3 v2i*)p;
}
__device__ __forceinline__ AS1 v4i* gp128(const void* p) {
    return (AS1 v4i*)(unsigned long long)p;
}
__device__ __forceinline__ AS3 v4i* sp128(void* p) {
    return (AS3 v4i*)p;
}
#endif

__device__ __forceinline__ void wait_async() {
#if HAVE_ASYNC_LDS
#if __has_builtin(__builtin_amdgcn_s_wait_asynccnt)
    __builtin_amdgcn_s_wait_asynccnt(0);
#else
    asm volatile("s_wait_asynccnt 0x0" ::: "memory");
#endif
#endif
}

__device__ __forceinline__ float gelu_exact(float v) {
    return 0.5f * v * (1.0f + erff(v * 0.70710678118654752f));
}
// order-preserving float<->int encoding (for atomicMax on float values)
__device__ __forceinline__ int enc_f(float f) {
    int i = __float_as_int(f);
    return i >= 0 ? i : (i ^ 0x7fffffff);
}
__device__ __forceinline__ float dec_f(int i) {
    return __int_as_float(i >= 0 ? i : (i ^ 0x7fffffff));
}

// ---- transpose f32 [K, Mc] -> bf16 [Mc, K] (weights; tiny, stays in L2) ----
__global__ void transpose_to_bf16(const float* __restrict__ src, __bf16* __restrict__ dst,
                                  int K, int Mc) {
    int idx = blockIdx.x * blockDim.x + threadIdx.x;
    if (idx >= K * Mc) return;
    int n = idx / K;
    int k = idx - n * K;
    dst[idx] = (__bf16)src[k * Mc + n];
}

// ---- edge gather (bf16) + scatter-add (f32 atomics): agg[row] += h[col] ----
__global__ __launch_bounds__(256) void scatter_add_kernel(
    const __bf16* __restrict__ h, const long long* __restrict__ ei,
    float* __restrict__ agg) {
    long long t = (long long)blockIdx.x * blockDim.x + threadIdx.x;
    int e = (int)(t >> 5);               // 32 lanes per edge
    if (e >= EE) return;
    int c0 = ((int)t & 31) * 8;          // 8 columns per lane
    long long r = ei[e];
    long long c = ei[EE + e];
    const __bf16* hp = h + c * MM + c0;
    float* ap = agg + r * MM + c0;
#pragma unroll
    for (int j = 0; j < 8; ++j)
        atomicAdd(ap + j, (float)hp[j]);
}

// ---- stage one 32xK_tile A block + 256xK_tile B block into LDS ----
template<int MODE>
__device__ __forceinline__ void stage_tiles(
    unsigned char* __restrict__ buf, int k0,
    long long row0, int tid,
    const float* __restrict__ A32, const __bf16* __restrict__ A16,
    const float* __restrict__ Aagg, const __bf16* __restrict__ WT, int K)
{
    __bf16* sA = (__bf16*)buf;
    __bf16* sB = sA + TILE_R * LDA;
    // ---- A tile 32x32 (8 threads per row, 4 elems per thread) ----
    {
        int r  = tid >> 3;
        int cs = (tid & 7) * 4;
        long long grow = row0 + r;
        __bf16* dst = sA + r * LDA + cs;
        const bool a_is_bf16 = (MODE == 2) || (MODE == 1 && k0 < MM); // uniform
        if (grow < NN) {
            if (a_is_bf16) {
                const __bf16* p = A16 + grow * MM + k0 + cs;
#if HAVE_ASYNC_LDS
                __builtin_amdgcn_global_load_async_to_lds_b64(gp64(p), sp64(dst), 0, 0);
#else
#pragma unroll
                for (int j = 0; j < 4; ++j) dst[j] = p[j];
#endif
            } else if (MODE == 0) {
                const float* p = A32 + grow * HIDK + k0 + cs;
#pragma unroll
                for (int j = 0; j < 4; ++j) dst[j] = (__bf16)p[j];
            } else { // MODE 1, agg half (f32 -> bf16 convert)
                const float* p = Aagg + grow * MM + (k0 - MM) + cs;
#pragma unroll
                for (int j = 0; j < 4; ++j) dst[j] = (__bf16)p[j];
            }
        } else {
#pragma unroll
            for (int j = 0; j < 4; ++j) dst[j] = (__bf16)0.f;
        }
    }
    // ---- B tile: WT[n, k0..k0+31] (K-contiguous bf16) -> sB[n][k] ----
    {
        const __bf16* p = WT + (long long)tid * K + k0;
        __bf16* dst = sB + tid * LDB;
#if HAVE_ASYNC_LDS
#pragma unroll
        for (int q = 0; q < 4; ++q)
            __builtin_amdgcn_global_load_async_to_lds_b128(
                gp128(p + q * 8), sp128(dst + q * 8), 0, 0);
#else
#pragma unroll
        for (int k = 0; k < 32; ++k) dst[k] = p[k];
#endif
    }
}

// ---- fused GEMM (bf16 WMMA, f32 accum) + bias + LayerNorm + exact GELU ----
// MODE 0: A = x (f32, K=512)         -> hout bf16
// MODE 1: A = [h bf16 | agg f32]     -> hout bf16   (K=512)
// MODE 2: A = h bf16 (K=256), epilogue -> per-block column sum/max partials
template<int MODE>
__global__ __launch_bounds__(256) void gemm_ln_gelu(
    const float* __restrict__ A32, const __bf16* __restrict__ A16,
    const float* __restrict__ Aagg, const __bf16* __restrict__ WT, // [MM, K] bf16
    const float* __restrict__ bias, const float* __restrict__ gamma,
    const float* __restrict__ beta, __bf16* __restrict__ hout,
    float* __restrict__ psum, int* __restrict__ pmax, int K)
{
    __shared__ __align__(16) unsigned char smem[SMEM_BYTES]; // 2 staging bufs / C tile
    __shared__ float s_sum[MM];
    __shared__ int   s_max[MM];

    float* sC = (float*)smem; // aliases staging buffers after final barrier

    const int tid  = threadIdx.x;
    const int lane = tid & 31;
    const int wid  = tid >> 5;       // 8 waves
    const int l16  = lane & 15;
    const int half = lane >> 4;
    const int wm   = wid & 1;        // 2 row groups of 16
    const int wn   = wid >> 1;       // 4 col groups of 64
    const long long row0 = (long long)blockIdx.x * TILE_R;

    if (MODE == 2 && tid < MM) { s_sum[tid] = 0.f; s_max[tid] = (int)0x80000000; }

    v8f acc[4] = {};

    // ---- software-pipelined K loop: prefetch next tile while WMMA-ing current ----
    stage_tiles<MODE>(smem, 0, row0, tid, A32, A16, Aagg, WT, K);
    wait_async();
    __syncthreads();

    const int nk = K / 32;
    for (int i = 0; i < nk; ++i) {
        const int cur = i & 1;
        __bf16* sAc = (__bf16*)(smem + cur * STG_BYTES);
        __bf16* sBc = sAc + TILE_R * LDA;
        if (i + 1 < nk)
            stage_tiles<MODE>(smem + (cur ^ 1) * STG_BYTES, (i + 1) * 32,
                              row0, tid, A32, A16, Aagg, WT, K);

        // ---- A fragment (ISA 16-bit A 16x32 layout) ----
        v16bf a;
        const __bf16* arow = sAc + (wm * 16 + l16) * LDA;
#pragma unroll
        for (int v = 0; v < 8; ++v) {
            int k = (v >> 2) * 16 + half * 8 + (v & 3) * 2;
            a[2 * v]     = arow[k];
            a[2 * v + 1] = arow[k + 1];
        }
        // ---- 4 WMMA tiles per wave (B: same K layout as A fragment) ----
#pragma unroll
        for (int t = 0; t < 4; ++t) {
            v16bf b;
            const __bf16* bcol = sBc + (wn * 64 + t * 16 + l16) * LDB;
#pragma unroll
            for (int v = 0; v < 8; ++v) {
                int k = (v >> 2) * 16 + half * 8 + (v & 3) * 2;
                b[2 * v]     = bcol[k];
                b[2 * v + 1] = bcol[k + 1];
            }
            acc[t] = __builtin_amdgcn_wmma_f32_16x16x32_bf16(
                false, a, false, b, (short)0, acc[t], false, false);
        }

        wait_async();        // next buffer's async copies complete
        __syncthreads();     // everyone done reading current buffer
    }

    // ---- epilogue: C (+bias) -> LDS (aliases staging region after barrier) ----
#pragma unroll
    for (int t = 0; t < 4; ++t) {
        int col = wn * 64 + t * 16 + l16;
        float bv = bias[col];
#pragma unroll
        for (int j = 0; j < 8; ++j) {
            int r = wm * 16 + half * 8 + j;
            sC[r * LDCE + col] = acc[t][j] + bv;
        }
    }
    __syncthreads();

    // ---- LayerNorm (biased var, as jnp.var) + exact GELU; 4 rows per wave ----
    for (int rr = 0; rr < 4; ++rr) {
        int r = wid * 4 + rr;
        long long grow = row0 + r;
        float s = 0.f, s2 = 0.f;
#pragma unroll
        for (int c = lane; c < MM; c += 32) {
            float v = sC[r * LDCE + c];
            s += v; s2 += v * v;
        }
#pragma unroll
        for (int o = 16; o > 0; o >>= 1) {
            s  += __shfl_xor(s, o, 32);
            s2 += __shfl_xor(s2, o, 32);
        }
        float mu   = s * (1.0f / MM);
        float var  = s2 * (1.0f / MM) - mu * mu;
        float rstd = rsqrtf(var + 1e-5f);
        if (grow < NN) {
#pragma unroll
            for (int c = lane; c < MM; c += 32) {
                float v = (sC[r * LDCE + c] - mu) * rstd * gamma[c] + beta[c];
                float y = gelu_exact(v);
                if (MODE < 2) {
                    hout[grow * MM + c] = (__bf16)y;
                } else {
                    atomicAdd(&s_sum[c], y);
                    atomicMax(&s_max[c], enc_f(y));
                }
            }
        }
    }
    if (MODE == 2) {
        __syncthreads();
        if (tid < MM) {
            psum[(long long)blockIdx.x * MM + tid] = s_sum[tid];
            pmax[(long long)blockIdx.x * MM + tid] = s_max[tid];
        }
    }
}

// ---- combine per-block partials -> out[0:256]=mean, out[256:512]=max ----
__global__ void finalize_kernel(const float* __restrict__ psum,
                                const int* __restrict__ pmax,
                                float* __restrict__ out, int nblocks) {
    int c = threadIdx.x;
    if (c >= MM) return;
    float s = 0.f; int m = (int)0x80000000;
    for (int b = 0; b < nblocks; ++b) {
        s += psum[(long long)b * MM + c];
        int v = pmax[(long long)b * MM + c];
        m = v > m ? v : m;
    }
    out[c]      = s * (1.0f / NN);
    out[MM + c] = dec_f(m);
}

extern "C" void kernel_launch(void* const* d_in, const int* in_sizes, int n_in,
                              void* d_out, int out_size, void* d_ws, size_t ws_size,
                              hipStream_t stream) {
    const float*     x       = (const float*)d_in[0];
    const long long* ei      = (const long long*)d_in[1];   // int64 [2, E]
    const float*     proj_W  = (const float*)d_in[2];
    const float*     proj_b  = (const float*)d_in[3];
    const float*     proj_g  = (const float*)d_in[4];
    const float*     proj_be = (const float*)d_in[5];
    const float*     agg_W   = (const float*)d_in[6];
    const float*     agg_b   = (const float*)d_in[7];
    const float*     agg_g   = (const float*)d_in[8];
    const float*     agg_be  = (const float*)d_in[9];
    const float*     out_W   = (const float*)d_in[10];
    const float*     out_b   = (const float*)d_in[11];
    const float*     out_g   = (const float*)d_in[12];
    const float*     out_be  = (const float*)d_in[13];
    float* out = (float*)d_out;

    char* ws = (char*)d_ws;
    size_t off = 0;
    auto alloc = [&](size_t bytes) {
        char* p = ws + off; off = (off + bytes + 255) & ~(size_t)255; return p;
    };
    __bf16* hA   = (__bf16*)alloc((size_t)NN * MM * 2);
    __bf16* hB   = (__bf16*)alloc((size_t)NN * MM * 2);
    float*  agg  = (float*) alloc((size_t)NN * MM * 4);
    __bf16* WTp  = (__bf16*)alloc((size_t)MM * HIDK * 2);
    __bf16* WTa  = (__bf16*)alloc((size_t)NSTEPS * MM * HIDK * 2);
    __bf16* WTo  = (__bf16*)alloc((size_t)MM * MM * 2);
    float*  psum = (float*) alloc((size_t)GBLK * MM * 4);
    int*    pmax = (int*)   alloc((size_t)GBLK * MM * 4);

    // pre-transpose weights to bf16 [N, K] (K-contiguous for WMMA B staging)
    {
        int tot = HIDK * MM;
        transpose_to_bf16<<<(tot + 255) / 256, 256, 0, stream>>>(proj_W, WTp, HIDK, MM);
        for (int s = 0; s < NSTEPS; ++s)
            transpose_to_bf16<<<(tot + 255) / 256, 256, 0, stream>>>(
                agg_W + (size_t)s * HIDK * MM, WTa + (size_t)s * MM * HIDK, HIDK, MM);
        int tot2 = MM * MM;
        transpose_to_bf16<<<(tot2 + 255) / 256, 256, 0, stream>>>(out_W, WTo, MM, MM);
    }

    // proj: x @ proj_W -> LN -> GELU -> hA (bf16)
    gemm_ln_gelu<0><<<GBLK, 256, 0, stream>>>(x, nullptr, nullptr, WTp,
        proj_b, proj_g, proj_be, hA, nullptr, nullptr, HIDK);

    __bf16* hcur = hA; __bf16* hnxt = hB;
    for (int s = 0; s < NSTEPS; ++s) {
        (void)hipMemsetAsync(agg, 0, (size_t)NN * MM * 4, stream);
        long long tthreads = (long long)EE * 32;
        scatter_add_kernel<<<(int)((tthreads + 255) / 256), 256, 0, stream>>>(hcur, ei, agg);
        gemm_ln_gelu<1><<<GBLK, 256, 0, stream>>>(nullptr, hcur, agg,
            WTa + (size_t)s * MM * HIDK,
            agg_b + s * MM, agg_g + s * MM, agg_be + s * MM,
            hnxt, nullptr, nullptr, 2 * MM);
        __bf16* t = hcur; hcur = hnxt; hnxt = t;
    }

    // out layer + per-block mean/max partials
    gemm_ln_gelu<2><<<GBLK, 256, 0, stream>>>(nullptr, hcur, nullptr, WTo,
        out_b, out_g, out_be, nullptr, psum, pmax, MM);

    finalize_kernel<<<1, 256, 0, stream>>>(psum, pmax, out, GBLK);
}
